// texture_feature_extractor_36661840839105
// MI455X (gfx1250) — matrix operand order repaired and checked by reading, not compiled, not run on hardware
//
#include <hip/hip_runtime.h>

#define FEAT 791
#define NB   100

typedef __attribute__((ext_vector_type(16))) _Float16 v16h;
typedef __attribute__((ext_vector_type(8)))  float    v8f;

// ---------- helpers ----------

__device__ __forceinline__ float gaussw(int ix, int iy, int n) {
  float cx = ((float)ix + 0.5f) / (float)n - 0.5f;
  float cy = ((float)iy + 0.5f) / (float)n - 0.5f;
  return __expf(-50.0f * (cx * cx + cy * cy));   // sigma=0.1 -> 1/(2s^2)=50
}

// mean of 2D gaussian window over n x n grid (separable)
__device__ __forceinline__ float mean_w(int n) {
  float s = 0.0f;
  for (int i = 0; i < n; ++i) {
    float c = ((float)i + 0.5f) / (float)n - 0.5f;
    s += __expf(-50.0f * c * c);
  }
  float m = s / (float)n;
  return m * m;
}

__device__ float breduce_sum(float v, float* red) {
  int tid = threadIdx.x;
  red[tid] = v; __syncthreads();
  for (int s = blockDim.x >> 1; s > 0; s >>= 1) {
    if (tid < s) red[tid] += red[tid + s];
    __syncthreads();
  }
  float r = red[0]; __syncthreads();
  return r;
}
__device__ float breduce_min(float v, float* red) {
  int tid = threadIdx.x;
  red[tid] = v; __syncthreads();
  for (int s = blockDim.x >> 1; s > 0; s >>= 1) {
    if (tid < s) red[tid] = fminf(red[tid], red[tid + s]);
    __syncthreads();
  }
  float r = red[0]; __syncthreads();
  return r;
}
__device__ float breduce_max(float v, float* red) {
  int tid = threadIdx.x;
  red[tid] = v; __syncthreads();
  for (int s = blockDim.x >> 1; s > 0; s >>= 1) {
    if (tid < s) red[tid] = fmaxf(red[tid], red[tid + s]);
    __syncthreads();
  }
  float r = red[0]; __syncthreads();
  return r;
}

// bbox half-width per reference: max(ceil(0.2*n), (out+1)/2), clamped
__device__ __forceinline__ int bbox_half(int n, int out_pix) {
  int half = (2 * n + 9) / 10;           // ceil(0.2*n)
  int mh = (out_pix + 1) / 2;
  if (half < mh) half = mh;
  int hmax = (n - 1) / 2;
  if (half > hmax) half = hmax;
  return half;
}

// ---------- weighted stats (mode0: m,v,s,k,min,max  mode1: v  mode2: s,k) ----------

__global__ void stats_kernel(const float* __restrict__ src, int n,
                             float* __restrict__ out, int off, int mode) {
  __shared__ float red[256];
  int b = blockIdx.x, tid = threadIdx.x;
  int N = n * n;
  const float* s = src + (size_t)b * N;
  float sw = 0.f, s1 = 0.f, s2 = 0.f, s3 = 0.f, s4 = 0.f;
  float mn = 3.4e38f, mx = -3.4e38f;
  for (int t = tid; t < N; t += blockDim.x) {
    int y = t / n, x = t % n;
    float w = gaussw(x, y, n);
    float v = s[t];
    float wv = w * v;
    sw += w; s1 += wv; s2 += wv * v; s3 += wv * v * v; s4 += wv * v * v * v;
    mn = fminf(mn, v); mx = fmaxf(mx, v);
  }
  sw = breduce_sum(sw, red);
  s1 = breduce_sum(s1, red);
  s2 = breduce_sum(s2, red);
  s3 = breduce_sum(s3, red);
  s4 = breduce_sum(s4, red);
  mn = breduce_min(mn, red);
  mx = breduce_max(mx, red);
  if (tid == 0) {
    float m = s1 / sw, e2 = s2 / sw, e3 = s3 / sw, e4 = s4 / sw;
    float v = e2 - m * m;
    float mu3 = e3 - 3.f * m * e2 + 2.f * m * m * m;
    float mu4 = e4 - 4.f * m * e3 + 6.f * m * m * e2 - 3.f * m * m * m * m;
    float sk = mu3 / (v * sqrtf(v));
    float ku = mu4 / (v * v);
    float* ob = out + (size_t)b * FEAT + off;
    if (mode == 0) { ob[0] = m; ob[1] = v; ob[2] = sk; ob[3] = ku; ob[4] = mn; ob[5] = mx; }
    else if (mode == 1) { ob[0] = v; }
    else { ob[0] = sk; ob[1] = ku; }
  }
}

// ---------- per (b,ff,o) means: mag, mag*w, re*w (+ coarse) ----------
// ws layout: ws[((b*4+ff)*4+o)*8 + slot]
// slots: 0 mean(mag) 1 mean(mag*w) 2 mean(re*w) 3 mean(cmag) 4 mean(cmag*w) 5 mean(cre*w) 6 mean(cim*w)

__global__ void means_kernel(const float* __restrict__ re, const float* __restrict__ im,
                             const float* __restrict__ cre, const float* __restrict__ cim,
                             int p, int ff, float* __restrict__ out, float* __restrict__ ws) {
  __shared__ float red[256];
  int o = blockIdx.x, b = blockIdx.y, tid = threadIdx.x;
  int N = p * p;
  size_t base = ((size_t)b * 4 + o) * N;
  const float* rr = re + base;
  const float* ii = im + base;
  float sm = 0.f, smw = 0.f, srw = 0.f, scm = 0.f, scmw = 0.f, scrw = 0.f, sciw = 0.f;
  for (int t = tid; t < N; t += blockDim.x) {
    int y = t / p, x = t % p;
    float w = gaussw(x, y, p);
    float a = rr[t], bb = ii[t];
    float mag = sqrtf(a * a + bb * bb);
    sm += mag; smw += mag * w; srw += a * w;
    if (cre) {
      float ca = cre[base + t], cb = cim[base + t];
      float cm = sqrtf(ca * ca + cb * cb);
      scm += cm; scmw += cm * w; scrw += ca * w; sciw += cb * w;
    }
  }
  sm = breduce_sum(sm, red);  smw = breduce_sum(smw, red);  srw = breduce_sum(srw, red);
  scm = breduce_sum(scm, red); scmw = breduce_sum(scmw, red);
  scrw = breduce_sum(scrw, red); sciw = breduce_sum(sciw, red);
  if (tid == 0) {
    float invN = 1.0f / (float)N;
    float mw = mean_w(p);
    float* wsb = ws + (((size_t)b * 4 + ff) * 4 + o) * 8;
    wsb[0] = sm * invN; wsb[1] = smw * invN; wsb[2] = srw * invN;
    wsb[3] = scm * invN; wsb[4] = scmw * invN; wsb[5] = scrw * invN; wsb[6] = sciw * invN;
    float* ob = out + (size_t)b * FEAT;
    ob[6 + ff * 4 + o]  = (smw * invN) / mw;   // mean_mag
    ob[22 + ff * 4 + o] = (srw * invN) / mw;   // mean_real
  }
}

// ---------- circular autocorrelation core ----------

__device__ void autocorr_body(float* patch, float* red, int side, int out_pix,
                              float* obase) {
  int tid = threadIdx.x, bs = blockDim.x;
  int S = side * side;
  // subtract patch mean
  float psum = 0.f;
  for (int t = tid; t < S; t += bs) psum += patch[t];
  psum = breduce_sum(psum, red);
  float mean = psum / (float)S;
  for (int t = tid; t < S; t += bs) patch[t] -= mean;
  __syncthreads();
  int nvals = (out_pix * out_pix + 1) / 2;
  int h = out_pix / 2;
  for (int lag = 0; lag < nvals; ++lag) {
    int dy = lag / out_pix - h;
    int dx = lag % out_pix - h;
    float acc = 0.f;
    for (int t = tid; t < S; t += bs) {
      int py = t / side, px = t % side;
      int qy = py + dy; if (qy < 0) qy += side; if (qy >= side) qy -= side;
      int qx = px + dx; if (qx < 0) qx += side; if (qx >= side) qx -= side;
      acc += patch[t] * patch[qy * side + qx];
    }
    acc = breduce_sum(acc, red);
    if (tid == 0) obase[lag] = acc;
    __syncthreads();
  }
  for (int t = nvals + tid; t < 25; t += bs) obase[t] = 0.0f;
}

__global__ void autocorr_plain(const float* __restrict__ src, int n, int out_pix,
                               int out_base, float* __restrict__ out) {
  __shared__ float patch[105 * 105];
  __shared__ float red[256];
  int b = blockIdx.x, tid = threadIdx.x, bs = blockDim.x;
  int half = bbox_half(n, out_pix);
  int r0 = n / 2 - half;
  int side = 2 * half + 1, S = side * side;
  const float* s = src + (size_t)b * n * n;
  for (int t = tid; t < S; t += bs) {
    int py = t / side, px = t % side;
    int y = r0 + py, x = r0 + px;
    patch[t] = s[y * n + x] * gaussw(x, y, n);
  }
  __syncthreads();
  autocorr_body(patch, red, side, out_pix, out + (size_t)b * FEAT + out_base);
}

__global__ void autocorr_mag(const float* __restrict__ re, const float* __restrict__ im,
                             int n, int ff, int out_pix, int out_base,
                             const float* __restrict__ ws, float* __restrict__ out) {
  __shared__ float patch[105 * 105];
  __shared__ float red[256];
  int o = blockIdx.x, b = blockIdx.y, tid = threadIdx.x, bs = blockDim.x;
  int half = bbox_half(n, out_pix);
  int r0 = n / 2 - half;
  int side = 2 * half + 1, S = side * side;
  size_t base = ((size_t)b * 4 + o) * n * n;
  float mmean = ws[(((size_t)b * 4 + ff) * 4 + o) * 8 + 0];
  for (int t = tid; t < S; t += bs) {
    int py = t / side, px = t % side;
    int y = r0 + py, x = r0 + px;
    float a = re[base + y * n + x], bb = im[base + y * n + x];
    patch[t] = (sqrtf(a * a + bb * bb) - mmean) * gaussw(x, y, n);
  }
  __syncthreads();
  autocorr_body(patch, red, side, out_pix,
                out + (size_t)b * FEAT + out_base + o * 25);
}

// ---------- WMMA Gram kernel: one wave per (image, scale) ----------
// Rows  A  = [wm0..3, wr0..3, 0,0]        (16 x K)
// Cols  B2 = [wcm0..3, wcr0..3, wci0..3,0](K x 16)
// G1 = A*A^T  -> mag_within / real_within ; G2 = A*B2 -> across terms

__global__ __launch_bounds__(32) void gram_kernel(
    const float* __restrict__ re, const float* __restrict__ im,
    const float* __restrict__ cre, const float* __restrict__ cim,
    int p, int ff, const float* __restrict__ ws, float* __restrict__ out) {
  int b = blockIdx.x;
  int lane = threadIdx.x;
  int np = p * p;
  size_t base4 = (size_t)b * 4 * np;

  __shared__ float M[4][7];
  if (lane < 28) {
    int o = lane / 7, s = lane % 7;
    M[o][s] = ws[(((size_t)b * 4 + ff) * 4 + o) * 8 + s];
  }
  float mw = mean_w(p);

  __shared__ _Float16 TA[16][32];
  __shared__ _Float16 TB[16][32];
  for (int ch = 0; ch < 16; ++ch) { TA[ch][lane] = (_Float16)0.f; TB[ch][lane] = (_Float16)0.f; }
  __syncthreads();

  v8f c1 = {}; v8f c2 = {};

  for (int kb = 0; kb < np; kb += 32) {
    int t = kb + lane;
    int y = t / p, x = t % p;
    float w = gaussw(x, y, p);
    if (t + 2048 < np) {  // streaming prefetch hints (global_prefetch)
      __builtin_prefetch(re + base4 + t + 2048, 0, 1);
      __builtin_prefetch(im + base4 + t + 2048, 0, 1);
    }
    for (int o = 0; o < 4; ++o) {
      float rr = re[base4 + (size_t)o * np + t];
      float ii = im[base4 + (size_t)o * np + t];
      float mag = sqrtf(rr * rr + ii * ii);
      float wm = (mag - M[o][0]) * w - (M[o][1] - M[o][0] * mw);
      float wr = rr * w - M[o][2];
      TA[o][lane] = (_Float16)wm;
      TA[4 + o][lane] = (_Float16)wr;
      if (ff > 0) {
        float ca = cre[base4 + (size_t)o * np + t];
        float cb = cim[base4 + (size_t)o * np + t];
        float cm = sqrtf(ca * ca + cb * cb);
        float wcm = (cm - M[o][3]) * w - (M[o][4] - M[o][3] * mw);
        float wcr = ca * w - M[o][5];
        float wci = cb * w - M[o][6];
        TB[o][lane] = (_Float16)wcm;
        TB[4 + o][lane] = (_Float16)wcr;
        TB[8 + o][lane] = (_Float16)wci;
      }
    }
    __syncthreads();

    // Build fragments with contiguous k order k = 16*kh + s, same in A and B (wave32).
    int mrow = lane & 15, kh = lane >> 4;
    union { v16h v; _Float16 h[16]; } A, B1, B2;
    for (int j = 0; j < 8; ++j) {
      int ka = 16 * kh + 2 * j;  // A: K map
      A.h[2 * j]     = TA[mrow][ka];
      A.h[2 * j + 1] = TA[mrow][ka + 1];
      int kb0 = 16 * kh + 2 * j;                                          // B: K map
      B1.h[2 * j]     = TA[mrow][kb0];
      B1.h[2 * j + 1] = TA[mrow][kb0 + 1];
      B2.h[2 * j]     = TB[mrow][kb0];
      B2.h[2 * j + 1] = TB[mrow][kb0 + 1];
    }
    c1 = __builtin_amdgcn_wmma_f32_16x16x32_f16(false, A.v, false, B1.v,
                                                (short)0, c1, false, false);
    if (ff > 0)
      c2 = __builtin_amdgcn_wmma_f32_16x16x32_f16(false, A.v, false, B2.v,
                                                  (short)0, c2, false, false);
    __syncthreads();
  }

  // C/D layout: lanes 0-15 hold N=lane, M=vgpr index (0..7). Rows >=8 unused.
  union { v8f v; float f[8]; } C1, C2;
  C1.v = c1; C2.v = c2;
  float inv = 1.0f / (float)np;
  float* ob = out + (size_t)b * FEAT;
  if (lane < 16) {
    const int P1[6] = {0, 0, 0, 1, 1, 2};
    const int P2[6] = {1, 2, 3, 2, 3, 3};
    for (int i = 0; i < 6; ++i) {
      if (lane == P2[i])     ob[599 + ff * 6 + i] = C1.f[P1[i]] * inv;        // mag_within
      if (lane == 4 + P2[i]) ob[623 + ff * 6 + i] = C1.f[4 + P1[i]] * inv;    // real_within
    }
    if (ff > 0) {
      if (lane < 4) {
        for (int o = 0; o < 4; ++o)
          ob[647 + (ff - 1) * 16 + o * 4 + lane] = C2.f[o] * inv;             // mag_across
      } else if (lane < 8) {
        int pp = lane - 4;
        for (int o = 0; o < 4; ++o)
          ob[695 + (ff - 1) * 32 + o * 4 + pp] = C2.f[4 + o] * inv;           // wr.wcr
      } else if (lane < 12) {
        int pp = lane - 8;
        for (int o = 0; o < 4; ++o)
          ob[695 + (ff - 1) * 32 + 16 + o * 4 + pp] = C2.f[4 + o] * inv;      // wr.wci
      }
    }
  }
}

// ---------- launch ----------

extern "C" void kernel_launch(void* const* d_in, const int* in_sizes, int n_in,
                              void* d_out, int out_size, void* d_ws, size_t ws_size,
                              hipStream_t stream) {
  (void)in_sizes; (void)n_in; (void)out_size; (void)ws_size;
  const float* images = (const float*)d_in[0];
  const float* c0r = (const float*)d_in[1];
  const float* c0i = (const float*)d_in[2];
  const float* c1r = (const float*)d_in[3];
  const float* c1i = (const float*)d_in[4];
  const float* c2r = (const float*)d_in[5];
  const float* c2i = (const float*)d_in[6];
  const float* c3r = (const float*)d_in[7];
  const float* c3i = (const float*)d_in[8];
  const float* o1r = (const float*)d_in[9];
  const float* o1i = (const float*)d_in[10];
  const float* o2r = (const float*)d_in[11];
  const float* o2i = (const float*)d_in[12];
  const float* o3r = (const float*)d_in[13];
  const float* o3i = (const float*)d_in[14];
  const float* resid_high = (const float*)d_in[16];
  const float* lp0 = (const float*)d_in[17];
  const float* lp1 = (const float*)d_in[18];
  const float* lp2 = (const float*)d_in[19];
  const float* lp3 = (const float*)d_in[20];
  const float* lp4 = (const float*)d_in[21];

  float* out = (float*)d_out;
  float* ws = (float*)d_ws;

  // 1) weighted pixel / residual / lowpass stats
  stats_kernel<<<NB, 256, 0, stream>>>(images, 256, out, 0, 0);     // pixel_stats
  stats_kernel<<<NB, 256, 0, stream>>>(resid_high, 256, out, 48, 1);// var_hp
  stats_kernel<<<NB, 256, 0, stream>>>(lp0, 16, out, 38, 2);        // marg
  stats_kernel<<<NB, 256, 0, stream>>>(lp1, 32, out, 40, 2);
  stats_kernel<<<NB, 256, 0, stream>>>(lp2, 64, out, 42, 2);
  stats_kernel<<<NB, 256, 0, stream>>>(lp3, 128, out, 44, 2);
  stats_kernel<<<NB, 256, 0, stream>>>(lp4, 256, out, 46, 2);

  // 2) per-(b,ff,o) means (feeds grams + mag autocorr; emits mean_mag/mean_real)
  dim3 g4(4, NB);
  means_kernel<<<g4, 256, 0, stream>>>(c0r, c0i, nullptr, nullptr, 32, 0, out, ws);
  means_kernel<<<g4, 256, 0, stream>>>(c1r, c1i, o1r, o1i, 64, 1, out, ws);
  means_kernel<<<g4, 256, 0, stream>>>(c2r, c2i, o2r, o2i, 128, 2, out, ws);
  means_kernel<<<g4, 256, 0, stream>>>(c3r, c3i, o3r, o3i, 256, 3, out, ws);

  // 3) autocorrelations
  autocorr_plain<<<NB, 256, 0, stream>>>(resid_high, 256, 7, 574, out); // hp_ac
  autocorr_plain<<<NB, 256, 0, stream>>>(lp0, 16, 3, 449, out);         // lp_ac
  autocorr_plain<<<NB, 256, 0, stream>>>(lp1, 32, 3, 474, out);
  autocorr_plain<<<NB, 256, 0, stream>>>(lp2, 64, 5, 499, out);
  autocorr_plain<<<NB, 256, 0, stream>>>(lp3, 128, 7, 524, out);
  autocorr_plain<<<NB, 256, 0, stream>>>(lp4, 256, 7, 549, out);
  autocorr_mag<<<g4, 256, 0, stream>>>(c0r, c0i, 32, 0, 3, 49, ws, out);  // mag_acs
  autocorr_mag<<<g4, 256, 0, stream>>>(c1r, c1i, 64, 1, 3, 149, ws, out);
  autocorr_mag<<<g4, 256, 0, stream>>>(c2r, c2i, 128, 2, 5, 249, ws, out);
  autocorr_mag<<<g4, 256, 0, stream>>>(c3r, c3i, 256, 3, 7, 349, ws, out);

  // 4) WMMA Gram pass: within/across correlations
  gram_kernel<<<NB, 32, 0, stream>>>(c0r, c0i, nullptr, nullptr, 32, 0, ws, out);
  gram_kernel<<<NB, 32, 0, stream>>>(c1r, c1i, o1r, o1i, 64, 1, ws, out);
  gram_kernel<<<NB, 32, 0, stream>>>(c2r, c2i, o2r, o2i, 128, 2, ws, out);
  gram_kernel<<<NB, 32, 0, stream>>>(c3r, c3i, o3r, o3i, 256, 3, ws, out);
}
